// PatchCoreSingle_83193516523905
// MI455X (gfx1250) — compile-verified, matrix operand order checked by reading, not run on hardware
//
#include <hip/hip_runtime.h>
#include <hip/hip_bf16.h>
#include <math.h>

typedef __attribute__((ext_vector_type(16))) _Float16 v16h;
typedef __attribute__((ext_vector_type(8)))  float    v8f;

#define WMMA_F16(a, b, c) \
  __builtin_amdgcn_wmma_f32_16x16x32_f16(false, (a), false, (b), (short)0, (c), false, false)

static __device__ __forceinline__ unsigned long long shflxor64(unsigned long long v, int m) {
  unsigned lo = (unsigned)(v & 0xFFFFFFFFull);
  unsigned hi = (unsigned)(v >> 32);
  lo = __shfl_xor(lo, m, 32);
  hi = __shfl_xor(hi, m, 32);
  return (((unsigned long long)hi) << 32) | (unsigned long long)lo;
}

static __device__ __forceinline__ unsigned long long umin64(unsigned long long a, unsigned long long b) {
  return a < b ? a : b;
}

union H4u { _Float16 h[4]; uint2 u; };
union F8u { v8f v; float f[8]; };

// ---------------------------------------------------------------------------
// init: reset per-patch (min d2, argmin) packed keys
// ---------------------------------------------------------------------------
__global__ void pc_init_kernel(unsigned long long* minkey, int P) {
  int p = blockIdx.x * blockDim.x + threadIdx.x;
  if (p < P) minkey[p] = 0xFFFFFFFFFFFFFFFFull;
}

// ---------------------------------------------------------------------------
// one-shot fp32 -> fp16 (hi) + fp16 residual (lo) conversion
// ---------------------------------------------------------------------------
__global__ void pc_convert_kernel(const float* __restrict__ x,
                                  _Float16* __restrict__ hi, _Float16* __restrict__ lo,
                                  int n4) {
  int i = blockIdx.x * blockDim.x + threadIdx.x;
  if (i >= n4) return;
  float4 v = ((const float4*)x)[i];
  float xa[4] = {v.x, v.y, v.z, v.w};
  H4u hh, hl;
#pragma unroll
  for (int j = 0; j < 4; ++j) {
    _Float16 h = (_Float16)xa[j];
    hh.h[j] = h;
    hl.h[j] = (_Float16)(xa[j] - (float)h);
  }
  ((uint2*)hi)[i] = hh.u;
  ((uint2*)lo)[i] = hl.u;
}

// ---------------------------------------------------------------------------
// row squared-norms (fp32 exact)
// ---------------------------------------------------------------------------
__global__ void pc_rownorm_kernel(const float* __restrict__ x, float* __restrict__ out,
                                  int rows, int cols) {
  __shared__ float ps[8];
  int row = blockIdx.x;
  if (row >= rows) return;
  const float4* xr = (const float4*)(x + (size_t)row * cols);
  int nf = cols >> 2;
  float s = 0.f;
  for (int f = threadIdx.x; f < nf; f += blockDim.x) {
    float4 v = xr[f];
    s += v.x * v.x + v.y * v.y + v.z * v.z + v.w * v.w;
  }
  for (int off = 16; off; off >>= 1) s += __shfl_xor(s, off, 32);
  if ((threadIdx.x & 31) == 0) ps[threadIdx.x >> 5] = s;
  __syncthreads();
  if (threadIdx.x == 0) {
    float t = 0.f;
    for (int i = 0; i < 8; ++i) t += ps[i];
    out[row] = t;
  }
}

// ---------------------------------------------------------------------------
// Fused GEMM (split fp16 hi/lo, 3 WMMAs per tile) + per-patch min/argmin.
// Block tile: 128(M=patches) x 128(N=bank rows), K stepped by 32.
// 256 threads = 8 waves; wave grid 4(M) x 2(N); each wave: 2x4 16x16 tiles.
// PRE=true: A/B already converted to fp16 hi/lo in workspace (pure copies).
// PRE=false: fallback converting fp32->fp16 in-kernel while staging.
// ---------------------------------------------------------------------------
static __device__ __forceinline__ v16h frag_ld(const _Float16* rowp, int kb) {
  union { v16h v; uint4 u[2]; } f;
  f.u[0] = *reinterpret_cast<const uint4*>(rowp + kb);
  f.u[1] = *reinterpret_cast<const uint4*>(rowp + kb + 16);
  return f.v;
}

template <bool PRE>
__global__ void __launch_bounds__(256)
pc_gemm_min_kernel(const float* __restrict__ Ag, const float* __restrict__ Bg,
                   const _Float16* __restrict__ Ah, const _Float16* __restrict__ Al,
                   const _Float16* __restrict__ Bh, const _Float16* __restrict__ Bl,
                   const float* __restrict__ a2, const float* __restrict__ b2,
                   unsigned long long* __restrict__ minkey,
                   int P, int N, int C) {
  __shared__ _Float16 sAhi[128][40];
  __shared__ _Float16 sAlo[128][40];
  __shared__ _Float16 sBhi[128][40];
  __shared__ _Float16 sBlo[128][40];
  __shared__ float sA2[128];
  __shared__ float sB2[128];

  const int tid  = threadIdx.x;
  const int wave = tid >> 5;
  const int lane = tid & 31;
  const int nl   = lane & 15;
  const int half = lane >> 4;
  const int wm = (wave & 3) * 32;   // 0,32,64,96
  const int wn = (wave >> 2) * 64;  // 0,64
  const int bm = blockIdx.y * 128;
  const int bn = blockIdx.x * 128;

  if (tid < 128) {
    sA2[tid] = a2[bm + tid];
  } else {
    int r = tid - 128;
    int n = bn + r;
    sB2[r] = (n < N) ? b2[n] : 0.f;
  }

  F8u acc[2][4];
#pragma unroll
  for (int mi = 0; mi < 2; ++mi)
#pragma unroll
    for (int t = 0; t < 4; ++t)
#pragma unroll
      for (int i = 0; i < 8; ++i) acc[mi][t].f[i] = 0.f;

  const int kb = half * 8;          // fragment K sub-offset per 16-lane half
  const int nk = C >> 5;

  for (int kt = 0; kt < nk; ++kt) {
    const int k0 = kt << 5;
    __syncthreads();
    if (PRE) {
      // Pure fp16 copies: 128x32 halves per tile component = 512 uint4.
#pragma unroll
      for (int i = 0; i < 2; ++i) {
        int f   = tid + i * 256;    // 0..511
        int row = f >> 2;
        int c8  = (f & 3) * 8;      // half-element offset within the 32-chunk
        size_t aoff = (size_t)(bm + row) * C + k0 + c8;
        *(uint4*)&sAhi[row][c8] = *(const uint4*)(Ah + aoff);
        *(uint4*)&sAlo[row][c8] = *(const uint4*)(Al + aoff);
        int nrow = bn + row;
        uint4 vh = make_uint4(0, 0, 0, 0), vl = vh;
        if (nrow < N) {
          size_t boff = (size_t)nrow * C + k0 + c8;
          vh = *(const uint4*)(Bh + boff);
          vl = *(const uint4*)(Bl + boff);
          if (kt + 1 < nk) {  // prefetch next K-chunk of the dominant B stream
            __builtin_prefetch(Bh + boff + 32, 0, 1);
            __builtin_prefetch(Bl + boff + 32, 0, 1);
          }
        }
        *(uint4*)&sBhi[row][c8] = vh;
        *(uint4*)&sBlo[row][c8] = vl;
      }
    } else {
      // Fallback: stage 128x32 fp32 -> fp16 hi/lo (A and B), 4 float4/thread
#pragma unroll
      for (int i = 0; i < 4; ++i) {
        int f   = tid + i * 256;    // 0..1023
        int row = f >> 3;
        int c4  = (f & 7) * 4;
        float4 va = *(const float4*)(Ag + (size_t)(bm + row) * C + k0 + c4);
        float xa[4] = {va.x, va.y, va.z, va.w};
        H4u ah, al;
#pragma unroll
        for (int j = 0; j < 4; ++j) {
          _Float16 h = (_Float16)xa[j];
          ah.h[j] = h;
          al.h[j] = (_Float16)(xa[j] - (float)h);
        }
        *(uint2*)&sAhi[row][c4] = ah.u;
        *(uint2*)&sAlo[row][c4] = al.u;
        int nrow = bn + row;
        float xb[4] = {0.f, 0.f, 0.f, 0.f};
        if (nrow < N) {
          float4 vb = *(const float4*)(Bg + (size_t)nrow * C + k0 + c4);
          xb[0] = vb.x; xb[1] = vb.y; xb[2] = vb.z; xb[3] = vb.w;
        }
        H4u bh, bl;
#pragma unroll
        for (int j = 0; j < 4; ++j) {
          _Float16 h = (_Float16)xb[j];
          bh.h[j] = h;
          bl.h[j] = (_Float16)(xb[j] - (float)h);
        }
        *(uint2*)&sBhi[row][c4] = bh.u;
        *(uint2*)&sBlo[row][c4] = bl.u;
      }
    }
    __syncthreads();

    v16h fbh[4], fbl[4];
#pragma unroll
    for (int t = 0; t < 4; ++t) {
      fbh[t] = frag_ld(&sBhi[wn + t * 16 + nl][0], kb);
      fbl[t] = frag_ld(&sBlo[wn + t * 16 + nl][0], kb);
    }
#pragma unroll
    for (int mi = 0; mi < 2; ++mi) {
      v16h fah = frag_ld(&sAhi[wm + mi * 16 + nl][0], kb);
      v16h fal = frag_ld(&sAlo[wm + mi * 16 + nl][0], kb);
      // split-precision: a.b ~= hi*hi + hi*lo + lo*hi  (fp32 accumulate)
#pragma unroll
      for (int t = 0; t < 4; ++t) {
        acc[mi][t].v = WMMA_F16(fah, fbh[t], acc[mi][t].v);
        acc[mi][t].v = WMMA_F16(fah, fbl[t], acc[mi][t].v);
        acc[mi][t].v = WMMA_F16(fal, fbh[t], acc[mi][t].v);
      }
    }
  }

  // Epilogue: d2 = a2 + b2 - 2*dot, clamp, pack (d2|n), min over N within wave
#pragma unroll
  for (int mi = 0; mi < 2; ++mi) {
#pragma unroll
    for (int r = 0; r < 8; ++r) {
      const int mloc = wm + mi * 16 + 8 * half + r;
      const float a2v = sA2[mloc];
      unsigned long long key = 0xFFFFFFFFFFFFFFFFull;
#pragma unroll
      for (int t = 0; t < 4; ++t) {
        const int nG = bn + wn + t * 16 + nl;
        if (nG < N) {
          float d2 = fmaxf(a2v + sB2[wn + t * 16 + nl] - 2.f * acc[mi][t].f[r], 0.f);
          unsigned long long k = (((unsigned long long)__float_as_uint(d2)) << 32) | (unsigned)nG;
          key = umin64(key, k);
        }
      }
      for (int m = 1; m < 16; m <<= 1) key = umin64(key, shflxor64(key, m));
      if (nl == 0) atomicMin(&minkey[bm + mloc], key);
    }
  }
}

// ---------------------------------------------------------------------------
// single block: dist_map = sqrt(min d2); argmax over patches (ties -> lowest p)
// ---------------------------------------------------------------------------
__global__ void pc_score_kernel(const unsigned long long* __restrict__ minkey,
                                float* __restrict__ out, unsigned* __restrict__ scal, int P) {
  __shared__ unsigned long long ps[8];
  unsigned long long best = 0ull;
  for (int p = threadIdx.x; p < P; p += 256) {
    unsigned long long k = minkey[p];
    unsigned d2b = (unsigned)(k >> 32);
    out[1 + p] = sqrtf(__uint_as_float(d2b));
    unsigned long long mk = (((unsigned long long)d2b) << 32) | (0xFFFFFFFFu - (unsigned)p);
    if (mk > best) best = mk;
  }
  for (int off = 16; off; off >>= 1) {
    unsigned long long o = shflxor64(best, off);
    if (o > best) best = o;
  }
  if ((threadIdx.x & 31) == 0) ps[threadIdx.x >> 5] = best;
  __syncthreads();
  if (threadIdx.x == 0) {
    unsigned long long b = ps[0];
    for (int i = 1; i < 8; ++i) if (ps[i] > b) b = ps[i];
    unsigned p = 0xFFFFFFFFu - (unsigned)(b & 0xFFFFFFFFull);
    scal[0] = __float_as_uint(sqrtf(__uint_as_float((unsigned)(b >> 32)))); // s_star
    scal[1] = p;                                          // s_idx
    scal[2] = (unsigned)(minkey[p] & 0xFFFFFFFFull);      // dist_idx[s_idx]
  }
}

// ---------------------------------------------------------------------------
// knn_d2[n] = ||bank[n] - m_star||^2 packed with index (one wave per row)
// ---------------------------------------------------------------------------
__global__ void pc_knn_kernel(const float* __restrict__ bank, const float* __restrict__ b2,
                              const unsigned* __restrict__ scal,
                              unsigned long long* __restrict__ knnkey, int N, int C) {
  int wave = threadIdx.x >> 5, lane = threadIdx.x & 31;
  int n = blockIdx.x * 8 + wave;
  if (n >= N) return;
  unsigned star = scal[2];
  const float4* ms = (const float4*)(bank + (size_t)star * C);
  const float4* bn_ = (const float4*)(bank + (size_t)n * C);
  float b2s = b2[star];
  int nf = C >> 2;
  float dot = 0.f;
  for (int f = lane; f < nf; f += 32) {
    float4 a = bn_[f], b = ms[f];
    dot += a.x * b.x + a.y * b.y + a.z * b.z + a.w * b.w;
  }
  for (int off = 16; off; off >>= 1) dot += __shfl_xor(dot, off, 32);
  if (lane == 0) {
    float d2 = fmaxf(b2[n] - 2.f * dot + b2s, 0.f);
    knnkey[n] = (((unsigned long long)__float_as_uint(d2)) << 32) | (unsigned)n;
  }
}

// ---------------------------------------------------------------------------
// single block: global top-3 smallest knn keys, then scalar math -> d_out[0]
// ---------------------------------------------------------------------------
static __device__ __forceinline__ void ins3(unsigned long long* t, unsigned long long k) {
  if (k < t[0])      { t[2] = t[1]; t[1] = t[0]; t[0] = k; }
  else if (k < t[1]) { t[2] = t[1]; t[1] = k; }
  else if (k < t[2]) { t[2] = k; }
}

__global__ void pc_final_kernel(const unsigned long long* __restrict__ knnkey,
                                const unsigned* __restrict__ scal,
                                const float* __restrict__ patches,
                                const float* __restrict__ bank,
                                float* __restrict__ out, int N, int C) {
  __shared__ unsigned long long tops[256 * 3];
  __shared__ unsigned nn[2];
  __shared__ float wd[2];
  __shared__ float ps[8];

  unsigned long long t3[3] = {0xFFFFFFFFFFFFFFFFull, 0xFFFFFFFFFFFFFFFFull, 0xFFFFFFFFFFFFFFFFull};
  for (int n = threadIdx.x; n < N; n += 256) ins3(t3, knnkey[n]);
  tops[threadIdx.x * 3 + 0] = t3[0];
  tops[threadIdx.x * 3 + 1] = t3[1];
  tops[threadIdx.x * 3 + 2] = t3[2];
  __syncthreads();
  if (threadIdx.x == 0) {
    unsigned long long g3[3] = {0xFFFFFFFFFFFFFFFFull, 0xFFFFFFFFFFFFFFFFull, 0xFFFFFFFFFFFFFFFFull};
    for (int i = 0; i < 256 * 3; ++i) ins3(g3, tops[i]);
    nn[0] = (unsigned)(g3[1] & 0xFFFFFFFFull);   // 2nd nearest (skip m_star itself)
    nn[1] = (unsigned)(g3[2] & 0xFFFFFFFFull);   // 3rd nearest
  }
  __syncthreads();

  const unsigned sidx = scal[1];
  const float4* pt = (const float4*)(patches + (size_t)sidx * C);
  const int nf = C >> 2;
  for (int j = 0; j < 2; ++j) {
    const float4* mb = (const float4*)(bank + (size_t)nn[j] * C);
    float s = 0.f;
    for (int f = threadIdx.x; f < nf; f += 256) {
      float4 a = pt[f], b = mb[f];
      float dx = a.x - b.x, dy = a.y - b.y, dz = a.z - b.z, dw = a.w - b.w;
      s += dx * dx + dy * dy + dz * dz + dw * dw;
    }
    for (int off = 16; off; off >>= 1) s += __shfl_xor(s, off, 32);
    if ((threadIdx.x & 31) == 0) ps[threadIdx.x >> 5] = s;
    __syncthreads();
    if (threadIdx.x == 0) {
      float t = 0.f;
      for (int i = 0; i < 8; ++i) t += ps[i];
      wd[j] = sqrtf(t);
    }
    __syncthreads();
  }
  if (threadIdx.x == 0) {
    float sstar = __uint_as_float(scal[0]);
    float nrm = sqrtf((float)C);
    float se = expf(wd[0] / nrm) + expf(wd[1] / nrm);
    float w = 1.f - expf(sstar / nrm) / se;
    out[0] = w * sstar;
  }
}

// ---------------------------------------------------------------------------
// launch
// ---------------------------------------------------------------------------
extern "C" void kernel_launch(void* const* d_in, const int* in_sizes, int n_in,
                              void* d_out, int out_size, void* d_ws, size_t ws_size,
                              hipStream_t stream) {
  (void)n_in; (void)out_size;
  const int C = 1536;
  const int P = in_sizes[0] / C;   // 4096
  const int N = in_sizes[1] / C;   // 50000
  const float* patches = (const float*)d_in[0];
  const float* bank    = (const float*)d_in[1];
  float* out = (float*)d_out;

  // workspace layout: small section (~650 KB) + optional fp16 hi/lo copies
  char* ws = (char*)d_ws;
  unsigned long long* minkey = (unsigned long long*)ws;                       // P * 8
  unsigned long long* knnkey = (unsigned long long*)(ws + (size_t)8 * P);     // N * 8
  float* b2 = (float*)(ws + (size_t)8 * P + (size_t)8 * N);                   // N * 4
  float* a2 = (float*)(ws + (size_t)8 * P + (size_t)12 * N);                  // P * 4
  unsigned* scal = (unsigned*)(ws + (size_t)12 * P + (size_t)12 * N);         // 16 * 4

  size_t base = (size_t)12 * P + (size_t)12 * N + 64;
  base = (base + 255) & ~(size_t)255;
  const size_t szA = (size_t)P * C * sizeof(_Float16);
  const size_t szB = (size_t)N * C * sizeof(_Float16);
  const bool pre = ws_size >= base + 2 * szA + 2 * szB;
  _Float16* Ah = (_Float16*)(ws + base);
  _Float16* Al = (_Float16*)(ws + base + szA);
  _Float16* Bh = (_Float16*)(ws + base + 2 * szA);
  _Float16* Bl = (_Float16*)(ws + base + 2 * szA + szB);

  pc_init_kernel<<<(P + 255) / 256, 256, 0, stream>>>(minkey, P);
  pc_rownorm_kernel<<<P, 256, 0, stream>>>(patches, a2, P, C);
  pc_rownorm_kernel<<<N, 256, 0, stream>>>(bank, b2, N, C);

  dim3 grid((N + 127) / 128, (P + 127) / 128);
  if (pre) {
    int a4 = P * C / 4, b4 = N * C / 4;
    pc_convert_kernel<<<(a4 + 255) / 256, 256, 0, stream>>>(patches, Ah, Al, a4);
    pc_convert_kernel<<<(b4 + 255) / 256, 256, 0, stream>>>(bank, Bh, Bl, b4);
    pc_gemm_min_kernel<true><<<grid, 256, 0, stream>>>(
        patches, bank, Ah, Al, Bh, Bl, a2, b2, minkey, P, N, C);
  } else {
    pc_gemm_min_kernel<false><<<grid, 256, 0, stream>>>(
        patches, bank, (const _Float16*)nullptr, (const _Float16*)nullptr,
        (const _Float16*)nullptr, (const _Float16*)nullptr, a2, b2, minkey, P, N, C);
  }

  pc_score_kernel<<<1, 256, 0, stream>>>(minkey, out, scal, P);
  pc_knn_kernel<<<(N + 7) / 8, 256, 0, stream>>>(bank, b2, scal, knnkey, N, C);
  pc_final_kernel<<<1, 256, 0, stream>>>(knnkey, scal, patches, bank, out, N, C);
}
